// MultiScaleNormalization_60447369724331
// MI455X (gfx1250) — matrix operand order repaired and checked by reading, not compiled, not run on hardware
//
#include <hip/hip_runtime.h>

typedef __attribute__((ext_vector_type(16))) __bf16 v16bf;
typedef __attribute__((ext_vector_type(8)))  __bf16 v8bf;
typedef __attribute__((ext_vector_type(8)))  float  v8f;
typedef __attribute__((ext_vector_type(4)))  unsigned int u32x4;
typedef __attribute__((ext_vector_type(4)))  int i32x4;
typedef __attribute__((ext_vector_type(8)))  int i32x8;

__device__ __forceinline__ float fast_rcp(float x) { return __builtin_amdgcn_rcpf(x); }
__device__ __forceinline__ float fast_rsq(float x) { return __builtin_amdgcn_rsqf(x); }
__device__ __forceinline__ float fast_silu(float v) { return v * fast_rcp(1.f + __expf(-v)); }

__device__ __forceinline__ v8f wmma_bf16(v16bf a, v16bf b, v8f c) {
  return __builtin_amdgcn_wmma_f32_16x16x32_bf16(
      false, a, false, b, (short)0, c, false, false);
}

// A-operand fragment: lane holds row m=l%16; K = 16*(l/16)+e, contiguous 16.
__device__ __forceinline__ v16bf load_a_frag(const __bf16* row, int off) {
  union { v16bf v; v8bf h[2]; } u;
  u.h[0] = *(const v8bf*)(row + 2 * off);
  u.h[1] = *(const v8bf*)(row + 2 * off + 8);
  return u.v;
}
// B-operand fragment: lane holds col n=l%16, K = 16*(l/16)+e, contiguous 16.
__device__ __forceinline__ v16bf load_b_frag(const __bf16* p) {
  union { v16bf v; v8bf h[2]; } u;
  u.h[0] = *(const v8bf*)(p);
  u.h[1] = *(const v8bf*)(p + 8);
  return u.v;
}

__device__ __forceinline__ v8bf cvt8(float4 a, float4 b) {
  v8bf r;
  r[0] = (__bf16)a.x; r[1] = (__bf16)a.y; r[2] = (__bf16)a.z; r[3] = (__bf16)a.w;
  r[4] = (__bf16)b.x; r[5] = (__bf16)b.y; r[6] = (__bf16)b.z; r[7] = (__bf16)b.w;
  return r;
}

// Tensor Data Mover: stage a 2D bf16 tile [rows x cols] (row-major, rowStride
// elements) from global into LDS (contiguous rows x cols). ISA 08 §8 D# layout.
__device__ __forceinline__ void tdm_load_tile_bf16(
    const __bf16* gsrc, unsigned ldsByteAddr, int cols, int rows,
    int tensorD0, int tensorD1, int rowStride) {
  const unsigned long long ga = (unsigned long long)(uintptr_t)gsrc;
  u32x4 g0;
  g0[0] = 1u;                                   // count=1, load, no gather
  g0[1] = ldsByteAddr;                          // lds_addr
  g0[2] = (unsigned)(ga & 0xFFFFFFFFull);       // global_addr[31:0]
  g0[3] = (unsigned)((ga >> 32) & 0x1FFFFFFull) | (2u << 30);  // addr[56:32]|type=2
  i32x8 g1;
  g1[0] = 0x10000;                              // data_size=1 (2 bytes)
  g1[1] = (int)((tensorD0 & 0xFFFF) << 16);     // tensor_dim0[15:0] (abar=0)
  g1[2] = (int)(((unsigned)tensorD0 >> 16) | ((unsigned)(tensorD1 & 0xFFFF) << 16));
  g1[3] = (int)(((unsigned)tensorD1 >> 16) | ((unsigned)cols << 16));  // tile_dim0
  g1[4] = rows;                                 // tile_dim1 (tile_dim2=0)
  g1[5] = rowStride;                            // tensor_dim0_stride[31:0]
  g1[6] = 0;
  g1[7] = 0;
  i32x4 z4 = {0, 0, 0, 0};
#if __has_include(<hip/amd_detail/amd_gfx1250_TDM.h>)
  i32x8 z8 = {0, 0, 0, 0, 0, 0, 0, 0};
  __builtin_amdgcn_tensor_load_to_lds(g0, g1, z4, z4, z8, 0);   // clang-23 arity
#else
  __builtin_amdgcn_tensor_load_to_lds(g0, g1, z4, z4, 0);       // ROCm 7.2 arity
#endif
}

// fp32 -> bf16 conversion (weights, done once per launch)
__global__ __launch_bounds__(256)
void cvt_bf16_kernel(const float* __restrict__ src, __bf16* __restrict__ dst, int n) {
  const int i = blockIdx.x * 256 + threadIdx.x;
  if (i < n) dst[i] = (__bf16)src[i];
}

// ---------------------------------------------------------------------------
// GEMM: C[M,N] = act(A[M,K] * Wb[N,K]^T + bias[N]).
// Block 256 threads (8 waves), 128x128 tile, waves 2x4, each 64x32 (4x2 WMMA).
// Double-buffered pipeline: TDM stages W tile k+1 and VMEM stages A tile k+1
// while WMMAs consume tile k; one barrier per K-step.
// Requires M%128==0, N%128==0, K%32==0 (true for all call sites).
// ---------------------------------------------------------------------------
#define BM 128
#define BN 128
#define BK 32

__device__ __forceinline__ void stage_a_tile(const void* A, int aBf16, int K,
                                             int mRow, int scol, int k0,
                                             __bf16* dst /* &As[buf][srow][scol] */,
                                             bool prefetchNext) {
  if (aBf16) {
    const __bf16* ap = (const __bf16*)A + (size_t)mRow * K + k0 + scol;
    if (prefetchNext) __builtin_prefetch(ap + BK, 0, 1);
    v8bf x0 = *(const v8bf*)(ap);
    v8bf x1 = *(const v8bf*)(ap + 8);
    *(v8bf*)(dst) = x0;
    *(v8bf*)(dst + 8) = x1;
  } else {
    const float* ap = (const float*)A + (size_t)mRow * K + k0 + scol;
    if (prefetchNext) __builtin_prefetch(ap + BK, 0, 1);
    float4 a0 = *(const float4*)(ap + 0);
    float4 a1 = *(const float4*)(ap + 4);
    float4 a2 = *(const float4*)(ap + 8);
    float4 a3 = *(const float4*)(ap + 12);
    *(v8bf*)(dst) = cvt8(a0, a1);
    *(v8bf*)(dst + 8) = cvt8(a2, a3);
  }
}

__global__ __launch_bounds__(256)
void gemm_bias_act(const void* __restrict__ A, int aBf16,
                   const __bf16* __restrict__ Wb, const float* __restrict__ bias,
                   void* __restrict__ C, int cBf16,
                   int M, int N, int K, int act) {
  __shared__ __align__(32) __bf16 As[2][BM][BK];
  __shared__ __align__(32) __bf16 Ws[2][BN][BK];

  const int tid = threadIdx.x;
  const int wave = tid >> 5, lane = tid & 31;
  const int laneLo = lane & 15, hi = lane >> 4;
  const int mBase = blockIdx.y * BM, nBase = blockIdx.x * BN;
  const int wm = (wave >> 2) * 64, wn = (wave & 3) * 32;
  const int off = hi ? 8 : 0;
  const bool issuer = (__builtin_amdgcn_readfirstlane((int)(tid >> 5)) == 0);
  const unsigned ldsWs0 = (unsigned)(uintptr_t)(&Ws[0][0][0]);
  const unsigned wsBufBytes = BN * BK * 2;      // 8 KB per buffer

  v8f acc[4][2];
#pragma unroll
  for (int i = 0; i < 4; ++i)
#pragma unroll
    for (int j = 0; j < 2; ++j) acc[i][j] = (v8f)0.f;

  const int srow = tid >> 1;            // 0..127
  const int scol = (tid & 1) * 16;      // 0 or 16
  const int mRow = mBase + srow;
  const __bf16* wTile = Wb + (size_t)nBase * K;
  const int nk = K / BK;

  // prologue: stage tile 0 into buffer 0
  if (issuer) tdm_load_tile_bf16(wTile, ldsWs0, BK, BN, K, N, K);
  stage_a_tile(A, aBf16, K, mRow, scol, 0, &As[0][srow][scol], nk > 1);
  if (issuer) __builtin_amdgcn_s_wait_tensorcnt(0);
  __syncthreads();

  for (int ki = 0; ki < nk; ++ki) {
    const int cur = ki & 1, nxt = cur ^ 1;

    // stage tile ki+1 into the other buffer (overlaps with WMMAs below)
    if (ki + 1 < nk) {
      if (issuer)
        tdm_load_tile_bf16(wTile + (ki + 1) * BK, ldsWs0 + nxt * wsBufBytes,
                           BK, BN, K, N, K);
      stage_a_tile(A, aBf16, K, mRow, scol, (ki + 1) * BK,
                   &As[nxt][srow][scol], ki + 2 < nk);
    }

    // compute on current buffer
    v16bf af[4], bf[2];
#pragma unroll
    for (int mt = 0; mt < 4; ++mt)
      af[mt] = load_a_frag(&As[cur][wm + mt * 16 + laneLo][0], off);
#pragma unroll
    for (int nt = 0; nt < 2; ++nt)
      bf[nt] = load_b_frag(&Ws[cur][wn + nt * 16 + laneLo][hi * 16]);
#pragma unroll
    for (int mt = 0; mt < 4; ++mt)
#pragma unroll
      for (int nt = 0; nt < 2; ++nt)
        acc[mt][nt] = wmma_bf16(af[mt], bf[nt], acc[mt][nt]);

    if (issuer && (ki + 1 < nk)) __builtin_amdgcn_s_wait_tensorcnt(0);
    __syncthreads();
  }

#pragma unroll
  for (int mt = 0; mt < 4; ++mt) {
#pragma unroll
    for (int nt = 0; nt < 2; ++nt) {
      const int gcol = nBase + wn + nt * 16 + laneLo;
      const float bn = bias[gcol];
#pragma unroll
      for (int r = 0; r < 8; ++r) {
        const int grow = mBase + wm + mt * 16 + r + hi * 8;
        float v = acc[mt][nt][r] + bn;
        if (act == 1) v = fast_silu(v);
        if (cBf16) ((__bf16*)C)[(size_t)grow * N + gcol] = (__bf16)v;
        else       ((float*)C)[(size_t)grow * N + gcol] = v;
      }
    }
  }
}

// ---------------------------------------------------------------------------
// LayerNorm (+ optional routing mask, + optional bf16 scatter to concat buf)
// ---------------------------------------------------------------------------
__global__ __launch_bounds__(256)
void ln_mask_kernel(const float* __restrict__ h, const float* __restrict__ g,
                    const float* __restrict__ beta, const int* __restrict__ types,
                    int eid, float* __restrict__ out, __bf16* __restrict__ cat,
                    int dim, int catStride, int catOff) {
  const int tok = blockIdx.x, tid = threadIdx.x;
  const float* row = h + (size_t)tok * dim;
  __shared__ float red[256];
  const float invD = 1.f / (float)dim;

  float s = 0.f;
  for (int i = tid; i < dim; i += 256) s += row[i];
  red[tid] = s; __syncthreads();
  for (int o = 128; o > 0; o >>= 1) { if (tid < o) red[tid] += red[tid + o]; __syncthreads(); }
  const float mu = red[0] * invD;
  __syncthreads();

  float vs = 0.f;
  for (int i = tid; i < dim; i += 256) { float d = row[i] - mu; vs += d * d; }
  red[tid] = vs; __syncthreads();
  for (int o = 128; o > 0; o >>= 1) { if (tid < o) red[tid] += red[tid + o]; __syncthreads(); }
  const float rstd = fast_rsq(red[0] * invD + 1e-5f);

  float mask = 1.f;
  if (types) mask = (types[tok] == eid) ? 1.f : 0.f;

  for (int i = tid; i < dim; i += 256) {
    const float v = (g[i] * (row[i] - mu) * rstd + beta[i]) * mask;
    out[(size_t)tok * dim + i] = v;
    if (cat) cat[(size_t)tok * catStride + catOff + i] = (__bf16)v;
  }
}

// ---------------------------------------------------------------------------
// Flash attention over bf16 QKV: block = (b,h,64-row q block); 4 waves x 16 q
// rows; kv chunks of 32, double-buffered so the next K/V fetch overlaps the
// Q*K^T WMMAs. o written bf16 head-concat.
// ---------------------------------------------------------------------------
#define DH 128
#define NHEADS 8

__global__ __launch_bounds__(128)
void attn_kernel(const __bf16* __restrict__ qkv, __bf16* __restrict__ o,
                 int B, int S, int D) {
  const int qb = blockIdx.x, h = blockIdx.y, b = blockIdx.z;
  const int tid = threadIdx.x;
  const int wave = tid >> 5, lane = tid & 31;
  const int laneLo = lane & 15, hi = lane >> 4;
  const int off = hi ? 8 : 0;

  __shared__ __align__(32) __bf16 Qs[64][DH];
  __shared__ __align__(32) __bf16 Ks[2][32][DH];
  __shared__ __align__(32) __bf16 Vts[2][DH][32];
  __shared__ __align__(32) __bf16 Ps[4][16][32];

  const float scale = 0.08838834764831845f;   // 1/sqrt(128)
  const size_t tok0 = (size_t)b * S + (size_t)qb * 64;
  const size_t trow = 3 * (size_t)D;
  const __bf16* qbase = qkv + tok0 * trow + (size_t)h * DH;
  const __bf16* kbase = qkv + (size_t)b * S * trow + D + (size_t)h * DH;
  const __bf16* vbase = kbase + D;

  // stage K/V chunk [j0, j0+32) into buffer buf.
  // 32*128 elems, 128 threads -> 4 v8bf (16B) loads each.
  auto stage_kv = [&](int j0, int buf) {
#pragma unroll
    for (int t = 0; t < 4; ++t) {
      const int idx = (tid + t * 128) * 8;      // element index
      const int r = idx / DH, c = idx % DH;
      const size_t goff = (size_t)(j0 + r) * trow + c;
      v8bf kv = *(const v8bf*)(kbase + goff);
      *(v8bf*)(&Ks[buf][r][c]) = kv;
      v8bf vv = *(const v8bf*)(vbase + goff);
#pragma unroll
      for (int e = 0; e < 8; ++e) Vts[buf][c + e][r] = vv[e];
    }
  };

  for (int idx = tid; idx < 64 * DH; idx += 128) {
    const int r = idx / DH, c = idx % DH;
    Qs[r][c] = (__bf16)((float)qbase[(size_t)r * trow + c] * scale);
  }
  stage_kv(0, 0);
  __syncthreads();

  v16bf qf[4];
#pragma unroll
  for (int ks = 0; ks < 4; ++ks)
    qf[ks] = load_a_frag(&Qs[wave * 16 + laneLo][ks * 32], off);

  v8f acc[8];
#pragma unroll
  for (int dt = 0; dt < 8; ++dt) acc[dt] = (v8f)0.f;
  float mstate[8], lstate[8];
#pragma unroll
  for (int r = 0; r < 8; ++r) { mstate[r] = -3.0e38f; lstate[r] = 0.f; }

  for (int j0 = 0; j0 < S; j0 += 32) {
    const int cur = (j0 >> 5) & 1, nxt = cur ^ 1;
    if (j0 + 32 < S) stage_kv(j0 + 32, nxt);    // overlaps with S-tile WMMAs

    v8f s0 = (v8f)0.f, s1 = (v8f)0.f;
#pragma unroll
    for (int ks = 0; ks < 4; ++ks) {
      v16bf b0 = load_b_frag(&Ks[cur][laneLo][ks * 32 + hi * 16]);
      v16bf b1 = load_b_frag(&Ks[cur][16 + laneLo][ks * 32 + hi * 16]);
      s0 = wmma_bf16(qf[ks], b0, s0);
      s1 = wmma_bf16(qf[ks], b1, s1);
    }

    float rowScale[8];
#pragma unroll
    for (int r = 0; r < 8; ++r) {
      float rm = fmaxf(s0[r], s1[r]);
#pragma unroll
      for (int d = 1; d < 16; d <<= 1) rm = fmaxf(rm, __shfl_xor(rm, d, 32));
      const float mnew = fmaxf(mstate[r], rm);
      const float sc = __expf(mstate[r] - mnew);
      const float p0 = __expf(s0[r] - mnew);
      const float p1 = __expf(s1[r] - mnew);
      float rs = p0 + p1;
#pragma unroll
      for (int d = 1; d < 16; d <<= 1) rs += __shfl_xor(rs, d, 32);
      lstate[r] = lstate[r] * sc + rs;
      mstate[r] = mnew;
      rowScale[r] = sc;
      const int rr = r + hi * 8;
      Ps[wave][rr][laneLo] = (__bf16)p0;
      Ps[wave][rr][16 + laneLo] = (__bf16)p1;
    }
#pragma unroll
    for (int dt = 0; dt < 8; ++dt)
#pragma unroll
      for (int r = 0; r < 8; ++r) acc[dt][r] *= rowScale[r];

    __syncthreads();   // Ps visible; next K/V staged

    const v16bf pf = load_a_frag(&Ps[wave][laneLo][0], off);
#pragma unroll
    for (int dt = 0; dt < 8; ++dt) {
      v16bf bv = load_b_frag(&Vts[cur][dt * 16 + laneLo][hi * 16]);
      acc[dt] = wmma_bf16(pf, bv, acc[dt]);
    }
    __syncthreads();   // cur/Ps reads done before they are overwritten
  }

  __bf16* obase = o + tok0 * D + (size_t)h * DH;
#pragma unroll
  for (int dt = 0; dt < 8; ++dt) {
#pragma unroll
    for (int r = 0; r < 8; ++r) {
      const int rr = r + hi * 8;
      const float ol = fast_rcp(lstate[r]);
      obase[(size_t)(wave * 16 + rr) * D + dt * 16 + laneLo] = (__bf16)(acc[dt][r] * ol);
    }
  }
}

// final = integrated + softmax(scale_w)[0] * attended
__global__ __launch_bounds__(256)
void final_add_kernel(const float* __restrict__ integ, const float* __restrict__ att,
                      const float* __restrict__ sw, float* __restrict__ out, size_t n) {
  const size_t i = (size_t)blockIdx.x * blockDim.x + threadIdx.x;
  const float m = fmaxf(fmaxf(sw[0], sw[1]), sw[2]);
  const float e0 = __expf(sw[0] - m), e1 = __expf(sw[1] - m), e2 = __expf(sw[2] - m);
  const float w0 = e0 * fast_rcp(e0 + e1 + e2);
  if (i < n) out[i] = integ[i] + w0 * att[i];
}

// ---------------------------------------------------------------------------
extern "C" void kernel_launch(void* const* d_in, const int* in_sizes, int n_in,
                              void* d_out, int out_size, void* d_ws, size_t ws_size,
                              hipStream_t stream) {
  const float* x     = (const float*)d_in[0];
  const int*   types = (const int*)d_in[1];
  const float* W[10] = { (const float*)d_in[2],  (const float*)d_in[4],   // mol W1,W2
                         (const float*)d_in[8],  (const float*)d_in[10],  // path
                         (const float*)d_in[14], (const float*)d_in[16],  // cell
                         (const float*)d_in[20], (const float*)d_in[22],  // int
                         (const float*)d_in[26], (const float*)d_in[28] };// qkv, wo
  const float* molb1  = (const float*)d_in[3],  *molb2  = (const float*)d_in[5];
  const float* molg   = (const float*)d_in[6],  *molbe  = (const float*)d_in[7];
  const float* pathb1 = (const float*)d_in[9],  *pathb2 = (const float*)d_in[11];
  const float* pathg  = (const float*)d_in[12], *pathbe = (const float*)d_in[13];
  const float* cellb1 = (const float*)d_in[15], *cellb2 = (const float*)d_in[17];
  const float* cellg  = (const float*)d_in[18], *cellbe = (const float*)d_in[19];
  const float* intb1  = (const float*)d_in[21], *intb2  = (const float*)d_in[23];
  const float* intg   = (const float*)d_in[24], *intbe  = (const float*)d_in[25];
  const float* bqkv   = (const float*)d_in[27], *bo     = (const float*)d_in[29];
  const float* sw     = (const float*)d_in[30];

  float* out = (float*)d_out;
  float* ws  = (float*)d_ws;

  const int S = 2048, D = 1024, B = 4, TOK = B * S;   // 8192
  const int CATD = 896;

  // fp32 workspace region (float offsets)
  const size_t OFF_H2  = 0;                           // 8192*1024
  const size_t OFF_INT = (size_t)TOK * D;             // 8388608
  const size_t OFF_ATT = 2 * (size_t)TOK * D;         // 16777216
  // bf16 region after 3 fp32 buffers
  __bf16* wsb = (__bf16*)(ws + 3 * (size_t)TOK * D);
  const size_t B_H1  = 0;                             // 8192*1024 (also o buf)
  const size_t B_CAT = (size_t)TOK * D;               // 8388608
  const size_t B_QKV = B_CAT + (size_t)TOK * CATD;    // 15728640 (8192*3072)
  const size_t B_W   = B_QKV + (size_t)TOK * 3 * D;   // weights bf16
  const size_t B_O   = B_H1;

  // per-weight bf16 offsets
  const int wsz[10] = { 128 * 1024, 128 * 128, 256 * 1024, 256 * 256,
                        512 * 1024, 512 * 512, 1024 * 896, 1024 * 1024,
                        3072 * 1024, 1024 * 1024 };
  size_t woff[10]; size_t acc = B_W;
  for (int i = 0; i < 10; ++i) { woff[i] = acc; acc += (size_t)wsz[i]; }
  for (int i = 0; i < 10; ++i)
    cvt_bf16_kernel<<<dim3((wsz[i] + 255) / 256), 256, 0, stream>>>(W[i], wsb + woff[i], wsz[i]);

  // d_out layout: final | mol | path | cell
  const size_t OUT_MOL  = (size_t)TOK * D;
  const size_t OUT_PATH = OUT_MOL + (size_t)TOK * 128;
  const size_t OUT_CELL = OUT_PATH + (size_t)TOK * 256;

  const dim3 blk256(256), blk128(128);
  const int gy = TOK / BM;                            // 64

  // --- mol expert ---
  gemm_bias_act<<<dim3(128 / BN, gy), blk256, 0, stream>>>(x, 0, wsb + woff[0], molb1, wsb + B_H1, 1, TOK, 128, D, 1);
  gemm_bias_act<<<dim3(128 / BN, gy), blk256, 0, stream>>>(wsb + B_H1, 1, wsb + woff[1], molb2, ws + OFF_H2, 0, TOK, 128, 128, 0);
  ln_mask_kernel<<<TOK, blk256, 0, stream>>>(ws + OFF_H2, molg, molbe, types, 0, out + OUT_MOL, wsb + B_CAT, 128, CATD, 0);

  // --- path expert ---
  gemm_bias_act<<<dim3(256 / BN, gy), blk256, 0, stream>>>(x, 0, wsb + woff[2], pathb1, wsb + B_H1, 1, TOK, 256, D, 1);
  gemm_bias_act<<<dim3(256 / BN, gy), blk256, 0, stream>>>(wsb + B_H1, 1, wsb + woff[3], pathb2, ws + OFF_H2, 0, TOK, 256, 256, 0);
  ln_mask_kernel<<<TOK, blk256, 0, stream>>>(ws + OFF_H2, pathg, pathbe, types, 1, out + OUT_PATH, wsb + B_CAT, 256, CATD, 128);

  // --- cell expert ---
  gemm_bias_act<<<dim3(512 / BN, gy), blk256, 0, stream>>>(x, 0, wsb + woff[4], cellb1, wsb + B_H1, 1, TOK, 512, D, 1);
  gemm_bias_act<<<dim3(512 / BN, gy), blk256, 0, stream>>>(wsb + B_H1, 1, wsb + woff[5], cellb2, ws + OFF_H2, 0, TOK, 512, 512, 0);
  ln_mask_kernel<<<TOK, blk256, 0, stream>>>(ws + OFF_H2, cellg, cellbe, types, 2, out + OUT_CELL, wsb + B_CAT, 512, CATD, 384);

  // --- integration MLP ---
  gemm_bias_act<<<dim3(D / BN, gy), blk256, 0, stream>>>(wsb + B_CAT, 1, wsb + woff[6], intb1, wsb + B_H1, 1, TOK, D, CATD, 1);
  gemm_bias_act<<<dim3(D / BN, gy), blk256, 0, stream>>>(wsb + B_H1, 1, wsb + woff[7], intb2, ws + OFF_H2, 0, TOK, D, D, 0);
  ln_mask_kernel<<<TOK, blk256, 0, stream>>>(ws + OFF_H2, intg, intbe, nullptr, -1, ws + OFF_INT, nullptr, D, 0, 0);

  // --- attention ---
  gemm_bias_act<<<dim3(3 * D / BN, gy), blk256, 0, stream>>>(ws + OFF_INT, 0, wsb + woff[8], bqkv, wsb + B_QKV, 1, TOK, 3 * D, D, 0);
  attn_kernel<<<dim3(S / 64, NHEADS, B), blk128, 0, stream>>>(wsb + B_QKV, wsb + B_O, B, S, D);
  gemm_bias_act<<<dim3(D / BN, gy), blk256, 0, stream>>>(wsb + B_O, 1, wsb + woff[9], bo, ws + OFF_ATT, 0, TOK, D, D, 0);

  // --- residual combine ---
  const size_t nFinal = (size_t)TOK * D;
  final_add_kernel<<<dim3((unsigned)((nFinal + 255) / 256)), blk256, 0, stream>>>(
      ws + OFF_INT, ws + OFF_ATT, sw, out, nFinal);
}
